// SSSDResidualLayer_43198781063855
// MI455X (gfx1250) — matrix operand order repaired and checked by reading, not compiled, not run on hardware
//
#include <hip/hip_runtime.h>

// ---------------------------------------------------------------------------
// SSSD residual layer for MI455X (gfx1250, wave32, WMMA).
#define BB 64
#define HH 256
#define LL 1024
#define NN 128
#define GG 8
#define HSS 256

typedef __attribute__((ext_vector_type(16))) __bf16 v16bf;
typedef __attribute__((ext_vector_type(8)))  __bf16 v8bf;
typedef __attribute__((ext_vector_type(8)))  float  v8f;
typedef __attribute__((ext_vector_type(4)))  float  v4f;
typedef __attribute__((ext_vector_type(4)))  unsigned int v4u;

__device__ __forceinline__ unsigned short f2bf(float f) {
    unsigned int u = __float_as_uint(f);
    unsigned int rnd = 0x7FFFu + ((u >> 16) & 1u);   // round-to-nearest-even
    return (unsigned short)((u + rnd) >> 16);
}

__device__ __forceinline__ float sigm(float x) { return 1.0f / (1.0f + __expf(-x)); }

__device__ __forceinline__ float gelu_tanh(float x) {
    const float k = 0.7978845608028654f; // sqrt(2/pi)
    float x3 = x * x * x;
    return 0.5f * x * (1.0f + tanhf(k * (x + 0.044715f * x3)));
}

// --- WMMA tile loaders (same contiguous k order 16h+s in A and B, wave32) ---------------
// A: 16x32 bf16 tile of row-major weight [O x K].  Per lane the needed K are
// one contiguous run: K = base..base+15 (base = 0 for lanes 0-15, 16 for
// lanes 16-31)  ->  two aligned 16-byte loads.
__device__ __forceinline__ v16bf load_A_bf16(const unsigned short* W, int ldk,
                                             int m0, int k0, int lane) {
    const unsigned short* row =
        W + (size_t)(m0 + (lane & 15)) * ldk + k0 + ((lane >> 4) ? 16 : 0);
    union { v16bf v; v8bf h[2]; } a;
    a.h[0] = *(const v8bf*)(row);
    a.h[1] = *(const v8bf*)(row + 8);
    return a.v;
}

// B: 32x16 bf16 tile with activations stored K-major ([L, H] rows).  Lane n
// (col) needs K = kb..kb+15 contiguous -> one aligned 32-byte load.
__device__ __forceinline__ v16bf load_B_LH(const unsigned short* Yt,
                                           int l0, int k0, int lane) {
    const unsigned short* ptr =
        Yt + (size_t)(l0 + (lane & 15)) * HH + k0 + ((lane >> 4) ? 16 : 0);
    return *(const v16bf*)ptr;
}

// ---------------------------------------------------------------------------
// K1: GroupNorm statistics. One block per (b,g); 32 ch x 1024 = 32768 floats.
__global__ void gn_stats_kernel(const float* __restrict__ x,
                                float* __restrict__ mu, float* __restrict__ rstd) {
    int bg = blockIdx.x;
    const float* base = x + (size_t)bg * 32 * LL;
    float s = 0.f, s2 = 0.f;
    for (int i = threadIdx.x; i < 32 * LL; i += 256) {
        float v = base[i];
        s += v; s2 += v * v;
    }
    __shared__ float sh1[256], sh2[256];
    sh1[threadIdx.x] = s; sh2[threadIdx.x] = s2;
    __syncthreads();
    for (int off = 128; off > 0; off >>= 1) {
        if (threadIdx.x < off) {
            sh1[threadIdx.x] += sh1[threadIdx.x + off];
            sh2[threadIdx.x] += sh2[threadIdx.x + off];
        }
        __syncthreads();
    }
    if (threadIdx.x == 0) {
        float inv = 1.0f / (32.0f * LL);
        float mean = sh1[0] * inv;
        float var = sh2[0] * inv - mean * mean;
        mu[bg] = mean;
        rstd[bg] = rsqrtf(var + 1e-5f);
    }
}

// K2: diffusion projection d[b,h] = emb[b,:]·dproj_w[h,:] + dproj_b[h]
__global__ void dproj_kernel(const float* __restrict__ emb, const float* __restrict__ W,
                             const float* __restrict__ bias, float* __restrict__ dvec) {
    int b = blockIdx.x, h = threadIdx.x;
    __shared__ float e[256];
    e[h] = emb[b * 256 + h];
    __syncthreads();
    const float* w = W + h * 256;
    float acc = bias[h];
#pragma unroll 8
    for (int k = 0; k < 256; ++k) acc += e[k] * w[k];
    dvec[b * HH + h] = acc;
}

// K3a: ZOH discretization: w = exp(dt*A), Ct = C*(w-1)/A    (H*N threads)
__global__ void setup_params_kernel(const float* __restrict__ log_dt,
                                    const float* __restrict__ A_re, const float* __restrict__ A_im,
                                    const float* __restrict__ C_re, const float* __restrict__ C_im,
                                    float* __restrict__ w_re, float* __restrict__ w_im,
                                    float* __restrict__ ct_re, float* __restrict__ ct_im) {
    int idx = blockIdx.x * 256 + threadIdx.x;   // H*N = 32768
    int h = idx >> 7;
    float dt = __expf(log_dt[h]);
    float ar = A_re[idx], ai = A_im[idx];
    float er = __expf(dt * ar);
    float sn, cs;
    __sincosf(dt * ai, &sn, &cs);
    float wr = er * cs, wi = er * sn;
    float emr = wr - 1.0f, emi = wi;            // exp(dtA) - 1
    float crr = C_re[idx], cii = C_im[idx];
    float nr = crr * emr - cii * emi;           // C * (exp(dtA)-1)
    float ni = crr * emi + cii * emr;
    float inv = 1.0f / (ar * ar + ai * ai);     // /A  (A never 0: re = -0.5)
    w_re[idx] = wr; w_im[idx] = wi;
    ct_re[idx] = (nr * ar + ni * ai) * inv;
    ct_im[idx] = (ni * ar - nr * ai) * inv;
}

// K3b: weights fp32 -> bf16
__global__ void convert_w_kernel(const float* __restrict__ out_w, const float* __restrict__ res_w,
                                 const float* __restrict__ skip_w,
                                 unsigned short* __restrict__ ob, unsigned short* __restrict__ rb,
                                 unsigned short* __restrict__ sb) {
    int i = blockIdx.x * 256 + threadIdx.x;     // 512 blocks -> 131072
    ob[i] = f2bf(out_w[i]);
    if (i < 256 * 256) {
        rb[i] = f2bf(res_w[i]);
        sb[i] = f2bf(skip_w[i]);
    }
}

// K4: fused GN-apply + diff-bias + diagonal SSM recurrence + D*u + GELU.
// One wave per (b,h) row; block = 8 consecutive channels of one batch.
// Outputs written K-major ([B, L, H]) via an LDS transpose tile so global
// stores are coalesced.
__global__ void ssm_kernel(const float* __restrict__ x,
                           const float* __restrict__ gn_w, const float* __restrict__ gn_b,
                           const float* __restrict__ mu, const float* __restrict__ rstd,
                           const float* __restrict__ dvec, const float* __restrict__ Dp,
                           const float* __restrict__ w_re, const float* __restrict__ w_im,
                           const float* __restrict__ ct_re, const float* __restrict__ ct_im,
                           float* __restrict__ Ut, unsigned short* __restrict__ Yt) {
    int wv = threadIdx.x >> 5, lane = threadIdx.x & 31;
    int b = blockIdx.x >> 5;              // 32 blocks per batch
    int h0 = (blockIdx.x & 31) * 8;
    int h = h0 + wv;
    int g = h >> 5;                       // 32 channels per group
    float m = mu[b * GG + g], rs = rstd[b * GG + g];
    float gw = gn_w[h] * rs;
    float gb = gn_b[h] - m * gw + dvec[b * HH + h];  // u = x*gw + gb
    float Dh = Dp[h];

    float wr[4], wi[4], cr[4], ci[4], sr[4], si[4];
    int nb = h * NN + lane * 4;
#pragma unroll
    for (int j = 0; j < 4; ++j) {
        wr[j] = w_re[nb + j]; wi[j] = w_im[nb + j];
        cr[j] = ct_re[nb + j]; ci[j] = ct_im[nb + j];
        sr[j] = 0.f; si[j] = 0.f;
    }
    const float* xr = x + ((size_t)b * HH + h) * LL;

    __shared__ float uT[32][9];                  // [l][h], pad 9 (9 coprime 64)
    __shared__ unsigned short yT[32][10];        // [l][h], padded

    for (int c = 0; c < LL / 32; ++c) {
        float u = xr[c * 32 + lane] * gw + gb;
        float yv = 0.f;
#pragma unroll 4
        for (int t = 0; t < 32; ++t) {
            float ut = __shfl(u, t, 32);
            float p = 0.f;
#pragma unroll
            for (int j = 0; j < 4; ++j) {        // s <- w*s + u  (complex diag)
                float nr2 = wr[j] * sr[j] - wi[j] * si[j] + ut;
                float ni2 = wr[j] * si[j] + wi[j] * sr[j];
                sr[j] = nr2; si[j] = ni2;
                p += cr[j] * sr[j] - ci[j] * si[j];
            }
            p += __shfl_xor(p, 16, 32);
            p += __shfl_xor(p, 8, 32);
            p += __shfl_xor(p, 4, 32);
            p += __shfl_xor(p, 2, 32);
            p += __shfl_xor(p, 1, 32);
            yv = (lane == t) ? p : yv;
        }
        float y = 2.0f * yv + Dh * u;
        uT[lane][wv] = u;
        yT[lane][wv] = f2bf(gelu_tanh(y));
        __syncthreads();
        // coalesced transposed stores: U 32B/8 threads, Y packed 32-bit
        {
            int row = threadIdx.x >> 3, hc = threadIdx.x & 7;
            Ut[((size_t)b * LL + c * 32 + row) * HH + h0 + hc] = uT[row][hc];
            if (threadIdx.x < 128) {
                int r2 = threadIdx.x >> 2, pr = threadIdx.x & 3;
                unsigned int pk = *(const unsigned int*)&yT[r2][2 * pr];
                *(unsigned int*)(Yt + ((size_t)b * LL + c * 32 + r2) * HH + h0 + 2 * pr) = pk;
            }
        }
        __syncthreads();
    }
}

// K5: out-projection (512x256) GEMM via WMMA bf16 + GLU + inner residual.
// One wave per paired 16x16 tile (rows m and m+256 share the B tile).
__global__ void outproj_glu_kernel(const unsigned short* __restrict__ Wb,   // [512,256] bf16
                                   const float* __restrict__ out_b,
                                   const unsigned short* __restrict__ Yt,   // [B,L,H] bf16
                                   const float* __restrict__ Ut,            // [B,L,H] f32
                                   unsigned short* __restrict__ HHt) {      // [B,L,H] bf16
    int wid = blockIdx.x * 8 + (threadIdx.x >> 5);
    int lane = threadIdx.x & 31;
    int lt = wid & 63, mt = (wid >> 6) & 15, b = wid >> 10;
    int m0 = mt * 16, l0 = lt * 16;
    const unsigned short* Yb = Yt + (size_t)b * LL * HH;

    v8f acc0 = {};  // value rows m0..m0+15
    v8f acc1 = {};  // gate  rows m0+256..
#pragma unroll
    for (int k0 = 0; k0 < 256; k0 += 32) {
        if (k0 + 32 < 256)
            __builtin_prefetch(Yb + (size_t)(l0 + (lane & 15)) * HH + k0 + 32, 0, 3);
        v16bf bt = load_B_LH(Yb, l0, k0, lane);
        v16bf a0 = load_A_bf16(Wb, 256, m0, k0, lane);
        v16bf a1 = load_A_bf16(Wb, 256, m0 + 256, k0, lane);
        acc0 = __builtin_amdgcn_wmma_f32_16x16x32_bf16(false, a0, false, bt,
                                                       (short)0, acc0, false, false);
        acc1 = __builtin_amdgcn_wmma_f32_16x16x32_bf16(false, a1, false, bt,
                                                       (short)0, acc1, false, false);
    }
    int col = l0 + (lane & 15);
    int c0 = m0 + ((lane >> 4) ? 8 : 0);      // 8 consecutive output channels
    const float* up = Ut + ((size_t)b * LL + col) * HH + c0;
    v4f u0 = *(const v4f*)(up);
    v4f u1 = *(const v4f*)(up + 4);
    v4f bt0 = *(const v4f*)(out_b + c0);
    v4f bt1 = *(const v4f*)(out_b + c0 + 4);
    v4f bg0 = *(const v4f*)(out_b + c0 + 256);
    v4f bg1 = *(const v4f*)(out_b + c0 + 260);
    union { v4u u4; unsigned short s[8]; } pk;
#pragma unroll
    for (int r = 0; r < 8; ++r) {
        float vtop = acc0[r] + ((r < 4) ? bt0[r & 3] : bt1[r & 3]);
        float vbot = acc1[r] + ((r < 4) ? bg0[r & 3] : bg1[r & 3]);
        float uu = (r < 4) ? u0[r & 3] : u1[r & 3];
        pk.s[r] = f2bf(vtop * sigm(vbot) + uu);   // GLU + inner residual
    }
    *(v4u*)(HHt + ((size_t)b * LL + col) * HH + c0) = pk.u4;  // 16B store
}

// K6: res & skip projections (both 256x256) share the B tile of hh.
// Epilogue transposes both tiles through LDS so each lane writes one
// contiguous 64B output row chunk in the required [B,H,L] layout.
__global__ void res_skip_kernel(const unsigned short* __restrict__ Rw,      // [256,256] bf16
                                const unsigned short* __restrict__ Sw,      // [256,256] bf16
                                const float* __restrict__ res_b, const float* __restrict__ skip_b,
                                const unsigned short* __restrict__ HHt,     // [B,L,H] bf16
                                const float* __restrict__ x,
                                float* __restrict__ out_res, float* __restrict__ out_skip) {
    int wv = threadIdx.x >> 5, lane = threadIdx.x & 31;
    int wid = blockIdx.x * 8 + wv;
    int lt = wid & 63, mt = (wid >> 6) & 15, b = wid >> 10;
    int m0 = mt * 16, l0 = lt * 16;
    const unsigned short* Hb = HHt + (size_t)b * LL * HH;

    v8f ar = {};
    v8f as = {};
#pragma unroll
    for (int k0 = 0; k0 < 256; k0 += 32) {
        if (k0 + 32 < 256)
            __builtin_prefetch(Hb + (size_t)(l0 + (lane & 15)) * HH + k0 + 32, 0, 3);
        v16bf bt = load_B_LH(Hb, l0, k0, lane);
        v16bf wa = load_A_bf16(Rw, 256, m0, k0, lane);
        v16bf wb = load_A_bf16(Sw, 256, m0, k0, lane);
        ar = __builtin_amdgcn_wmma_f32_16x16x32_bf16(false, wa, false, bt,
                                                     (short)0, ar, false, false);
        as = __builtin_amdgcn_wmma_f32_16x16x32_bf16(false, wb, false, bt,
                                                     (short)0, as, false, false);
    }
    __shared__ float lds[8][2][16][17];
    float (*tr)[17] = lds[wv][0];
    float (*ts)[17] = lds[wv][1];
    int nl = lane & 15, rb = (lane >> 4) ? 8 : 0;
#pragma unroll
    for (int r = 0; r < 8; ++r) {
        tr[rb + r][nl] = ar[r];
        ts[rb + r][nl] = as[r];
    }
    // wave-private produce->consume through LDS: wait on the DS counter
    asm volatile("s_wait_dscnt 0" ::: "memory");

    const float inv_sqrt2 = 0.70710678118654752f;
    int row = lane & 15;
    int c = m0 + row;
    if (lane < 16) {                       // lanes 0-15: one res row each
        const float* xp = x + ((size_t)b * HH + c) * LL + l0;
        float* op = out_res + ((size_t)b * HH + c) * LL + l0;
        float bias = res_b[c];
#pragma unroll
        for (int j = 0; j < 16; j += 4) {
            v4f xv = *(const v4f*)(xp + j);
            v4f o;
#pragma unroll
            for (int q = 0; q < 4; ++q)
                o[q] = (tr[row][j + q] + bias + xv[q]) * inv_sqrt2;
            *(v4f*)(op + j) = o;
        }
    } else {                               // lanes 16-31: one skip row each
        float* op = out_skip + ((size_t)b * HSS + c) * LL + l0;
        float bias = skip_b[c];
#pragma unroll
        for (int j = 0; j < 16; j += 4) {
            v4f o;
#pragma unroll
            for (int q = 0; q < 4; ++q)
                o[q] = ts[row][j + q] + bias;
            *(v4f*)(op + j) = o;
        }
    }
}

// ---------------------------------------------------------------------------
extern "C" void kernel_launch(void* const* d_in, const int* in_sizes, int n_in,
                              void* d_out, int out_size, void* d_ws, size_t ws_size,
                              hipStream_t stream) {
    const float* x        = (const float*)d_in[0];
    const float* demb     = (const float*)d_in[1];
    const float* gn_w     = (const float*)d_in[2];
    const float* gn_b     = (const float*)d_in[3];
    const float* dproj_w  = (const float*)d_in[4];
    const float* dproj_b  = (const float*)d_in[5];
    const float* log_dt   = (const float*)d_in[6];
    const float* A_re     = (const float*)d_in[7];
    const float* A_im     = (const float*)d_in[8];
    const float* C_re     = (const float*)d_in[9];
    const float* C_im     = (const float*)d_in[10];
    const float* Dp       = (const float*)d_in[11];
    const float* out_w    = (const float*)d_in[12];
    const float* out_b    = (const float*)d_in[13];
    const float* res_w    = (const float*)d_in[14];
    const float* res_b    = (const float*)d_in[15];
    const float* skip_w   = (const float*)d_in[16];
    const float* skip_b   = (const float*)d_in[17];
    (void)in_sizes; (void)n_in; (void)out_size; (void)ws_size;

    float* out_res  = (float*)d_out;
    float* out_skip = (float*)d_out + (size_t)BB * HH * LL;

    // Workspace carve-up (fp32 blocks first, then bf16/ushort blocks).
    char* p = (char*)d_ws;
    float* mu    = (float*)p;                 p += (size_t)BB * GG * 4;
    float* rstd  = (float*)p;                 p += (size_t)BB * GG * 4;
    float* dvec  = (float*)p;                 p += (size_t)BB * HH * 4;
    float* w_re  = (float*)p;                 p += (size_t)HH * NN * 4;
    float* w_im  = (float*)p;                 p += (size_t)HH * NN * 4;
    float* ct_re = (float*)p;                 p += (size_t)HH * NN * 4;
    float* ct_im = (float*)p;                 p += (size_t)HH * NN * 4;
    float* Ut    = (float*)p;                 p += (size_t)BB * HH * LL * 4;   // [B,L,H]
    unsigned short* out_wb = (unsigned short*)p; p += (size_t)2 * HH * HH * 2;
    unsigned short* res_wb = (unsigned short*)p; p += (size_t)HH * HH * 2;
    unsigned short* skp_wb = (unsigned short*)p; p += (size_t)HSS * HH * 2;
    unsigned short* Yt     = (unsigned short*)p; p += (size_t)BB * HH * LL * 2; // [B,L,H]
    unsigned short* HHt    = (unsigned short*)p; p += (size_t)BB * HH * LL * 2; // [B,L,H]

    gn_stats_kernel<<<BB * GG, 256, 0, stream>>>(x, mu, rstd);
    dproj_kernel<<<BB, 256, 0, stream>>>(demb, dproj_w, dproj_b, dvec);
    setup_params_kernel<<<(HH * NN) / 256, 256, 0, stream>>>(
        log_dt, A_re, A_im, C_re, C_im, w_re, w_im, ct_re, ct_im);
    convert_w_kernel<<<512, 256, 0, stream>>>(out_w, res_w, skip_w,
                                              out_wb, res_wb, skp_wb);
    ssm_kernel<<<(BB * HH) / 8, 256, 0, stream>>>(
        x, gn_w, gn_b, mu, rstd, dvec, Dp, w_re, w_im, ct_re, ct_im, Ut, Yt);
    outproj_glu_kernel<<<(BB * 16 * 64) / 8, 256, 0, stream>>>(
        out_wb, out_b, Yt, Ut, HHt);
    res_skip_kernel<<<(BB * 16 * 64) / 8, 256, 0, stream>>>(
        res_wb, skp_wb, res_b, skip_b, HHt, x, out_res, out_skip);
}
